// ViT_70394513981908
// MI455X (gfx1250) — matrix operand order repaired and checked by reading, not compiled, not run on hardware
//
#include <hip/hip_runtime.h>
#include <hip/hip_bf16.h>

#define DEV __device__ __forceinline__

typedef __bf16 v16bf __attribute__((ext_vector_type(16)));
typedef __bf16 v8bf  __attribute__((ext_vector_type(8)));
typedef __bf16 v4bf  __attribute__((ext_vector_type(4)));
typedef float  v8f   __attribute__((ext_vector_type(8)));
typedef float  v4f   __attribute__((ext_vector_type(4)));

#define NB     32
#define SEQL   197
#define DIMD   768
#define NHEADS 12
#define HDIM   64
#define MLPD   3072
#define NROWS  (NB * SEQL)     /* 6304  */
#define PROWS  (NB * 196)      /* 6272  */
#define SPAD   208             /* 197 -> 13 tiles of 16 */
#define KPAD   224             /* 197 -> 7 k-steps of 32 */
#define PSTR   448             /* bf16 stride of in-place probs over f32 scores */
#define BH     (NB * NHEADS)   /* 384 */

DEV __bf16 f2bf(float f) {
  unsigned u = __float_as_uint(f);
  unsigned r = (u + 0x7FFFu + ((u >> 16) & 1u)) >> 16;
  unsigned short s = (unsigned short)r;
  return __builtin_bit_cast(__bf16, s);
}

DEV v8f vzero8() {
  v8f r;
#pragma unroll
  for (int i = 0; i < 8; ++i) r[i] = 0.0f;
  return r;
}

// A fragment (16x32 bf16): lane holds row m=lane&15; 16 contiguous K values
// starting at k0 + 16*half ('half*16' folded into the pointer).
DEV v16bf load_a(const __bf16* arow) {
  v8bf lo = *(const v8bf*)(arow);
  v8bf hi = *(const v8bf*)(arow + 8);
  return __builtin_shufflevector(lo, hi, 0,1,2,3,4,5,6,7,8,9,10,11,12,13,14,15);
}

// B fragment: lane holds col n=lane&15, 16 contiguous K values starting at
// k0 + 16*half ('half*16' folded into the pointer).
DEV v16bf load_b16(const __bf16* wrow) {
  v8bf lo = *(const v8bf*)(wrow);
  v8bf hi = *(const v8bf*)(wrow + 8);
  return __builtin_shufflevector(lo, hi, 0,1,2,3,4,5,6,7,8,9,10,11,12,13,14,15);
}

DEV v8f wmma_bf16(v16bf a, v16bf b, v8f c) {
  return __builtin_amdgcn_wmma_f32_16x16x32_bf16(false, a, false, b, (short)0, c,
                                                 false, false);
}

// ---------------------------------------------------------------------------
// Generic GEMM: C[M,N] = A[M,K](bf16) * W[N,K]^T(bf16, pre-converted) + bias
// Wave computes a 32x64 tile: 2 A-frags, 4 B-frags, 8 WMMAs per K-step.
// Software-pipelined: next step's 6 fragment loads are issued before the
// current step's WMMAs so load latency hides behind matrix ops.
// EP: 0 f32+bias | 1 bf16+bias | 2 gelu->bf16 | 3 residual += | 4 patch scatter
//     5 head (clamped loads, col<Nreal store guard)
// ---------------------------------------------------------------------------
template <int EP>
__global__ __launch_bounds__(256) void gemm_aw(
    const __bf16* __restrict__ A, int lda, const __bf16* __restrict__ W, int ldw,
    const float* __restrict__ bias, float* __restrict__ Cf,
    __bf16* __restrict__ Cb, int ldc, int Mtiles /*M/32*/, int Ngroups /*N/64*/,
    int Nreal, int Ksteps, const float* __restrict__ pos) {
  const int wv = threadIdx.x >> 5, lane = threadIdx.x & 31;
  const int gw = blockIdx.x * 8 + wv;
  const int mt = gw / Ngroups, nt = gw % Ngroups;
  if (mt >= Mtiles) return;
  const int m0 = mt * 32, n0 = nt * 64;
  const int hf = lane >> 4, ml = lane & 15;

  const __bf16* ap0 = A + (size_t)(m0 + ml) * lda + hf * 16;
  const __bf16* ap1 = ap0 + (size_t)16 * lda;
  int nn0 = n0 + 0 * 16 + ml, nn1 = n0 + 1 * 16 + ml;
  int nn2 = n0 + 2 * 16 + ml, nn3 = n0 + 3 * 16 + ml;
  if (EP == 5) {  // clamp (stores guarded below)
    nn0 = min(nn0, Nreal - 1); nn1 = min(nn1, Nreal - 1);
    nn2 = min(nn2, Nreal - 1); nn3 = min(nn3, Nreal - 1);
  }
  const __bf16* wp0 = W + (size_t)nn0 * ldw + hf * 16;
  const __bf16* wp1 = W + (size_t)nn1 * ldw + hf * 16;
  const __bf16* wp2 = W + (size_t)nn2 * ldw + hf * 16;
  const __bf16* wp3 = W + (size_t)nn3 * ldw + hf * 16;

  v8f acc[8];
#pragma unroll
  for (int i = 0; i < 8; ++i) acc[i] = vzero8();

  // pipeline prologue
  v16bf a0 = load_a(ap0), a1 = load_a(ap1);
  v16bf b0 = load_b16(wp0), b1 = load_b16(wp1);
  v16bf b2 = load_b16(wp2), b3 = load_b16(wp3);

  for (int ks = 0; ks < Ksteps - 1; ++ks) {
    ap0 += 32; ap1 += 32;
    wp0 += 32; wp1 += 32; wp2 += 32; wp3 += 32;
    v16bf na0 = load_a(ap0), na1 = load_a(ap1);
    v16bf nb0 = load_b16(wp0), nb1 = load_b16(wp1);
    v16bf nb2 = load_b16(wp2), nb3 = load_b16(wp3);
    acc[0] = wmma_bf16(a0, b0, acc[0]);
    acc[4] = wmma_bf16(a1, b0, acc[4]);
    acc[1] = wmma_bf16(a0, b1, acc[1]);
    acc[5] = wmma_bf16(a1, b1, acc[5]);
    acc[2] = wmma_bf16(a0, b2, acc[2]);
    acc[6] = wmma_bf16(a1, b2, acc[6]);
    acc[3] = wmma_bf16(a0, b3, acc[3]);
    acc[7] = wmma_bf16(a1, b3, acc[7]);
    a0 = na0; a1 = na1;
    b0 = nb0; b1 = nb1; b2 = nb2; b3 = nb3;
  }
  acc[0] = wmma_bf16(a0, b0, acc[0]);
  acc[4] = wmma_bf16(a1, b0, acc[4]);
  acc[1] = wmma_bf16(a0, b1, acc[1]);
  acc[5] = wmma_bf16(a1, b1, acc[5]);
  acc[2] = wmma_bf16(a0, b2, acc[2]);
  acc[6] = wmma_bf16(a1, b2, acc[6]);
  acc[3] = wmma_bf16(a0, b3, acc[3]);
  acc[7] = wmma_bf16(a1, b3, acc[7]);

#pragma unroll
  for (int hm = 0; hm < 2; ++hm) {
    const int mb = m0 + hm * 16;
#pragma unroll
    for (int i = 0; i < 4; ++i) {
      const int col = n0 + i * 16 + ml;
      if (EP == 5 && col >= Nreal) continue;
      const float bv = bias[col];
#pragma unroll
      for (int r = 0; r < 8; ++r) {
        const int row = mb + hf * 8 + r;
        float v = acc[hm * 4 + i][r] + bv;
        if (EP == 0 || EP == 5) {
          Cf[(size_t)row * ldc + col] = v;
        } else if (EP == 1) {
          Cb[(size_t)row * ldc + col] = f2bf(v);
        } else if (EP == 2) {
          float g = 0.5f * v * (1.0f + erff(v * 0.70710678118f));
          Cb[(size_t)row * ldc + col] = f2bf(g);
        } else if (EP == 3) {
          Cf[(size_t)row * ldc + col] += v;
        } else if (EP == 4) {
          int bb = row / 196;     // batch
          int ro = row + bb + 1;  // skip CLS rows
          int s = ro - bb * SEQL; // token index (1..196)
          Cf[(size_t)ro * DIMD + col] = v + pos[(size_t)s * DIMD + col];
        }
      }
    }
  }
}

// ---------------------------------------------------------------------------
// Attention kernels
// ---------------------------------------------------------------------------
__global__ __launch_bounds__(128) void attn_qk(const __bf16* __restrict__ qkv,
                                               float* __restrict__ scores) {
  const int wv = threadIdx.x >> 5, lane = threadIdx.x & 31;
  const int bh = blockIdx.x, mt = blockIdx.y;
  const int b = bh / NHEADS, h = bh % NHEADS;
  const int hf = lane >> 4, ml = lane & 15;
  const int s = mt * 16 + ml;
  const int sc = (s < SEQL) ? s : (SEQL - 1);  // clamp pad rows (discarded)
  const __bf16* arow = qkv + (size_t)(b * SEQL + sc) * 2304 + h * HDIM + hf * 16;
  const int n0 = wv * 64;
  const __bf16* wp[4];
#pragma unroll
  for (int i = 0; i < 4; ++i) {
    int t = n0 + i * 16 + ml;
    t = (t < SEQL) ? t : (SEQL - 1);  // clamp: pad cols never read by softmax
    wp[i] = qkv + (size_t)(b * SEQL + t) * 2304 + 768 + h * HDIM + hf * 16;
  }
  // K = 64 -> 2 steps; fully straight-line: all loads first, then all WMMAs.
  v16bf aA = load_a(arow), aB = load_a(arow + 32);
  v16bf bA0 = load_b16(wp[0]), bA1 = load_b16(wp[1]);
  v16bf bA2 = load_b16(wp[2]), bA3 = load_b16(wp[3]);
  v16bf bB0 = load_b16(wp[0] + 32), bB1 = load_b16(wp[1] + 32);
  v16bf bB2 = load_b16(wp[2] + 32), bB3 = load_b16(wp[3] + 32);
  v8f acc[4];
#pragma unroll
  for (int i = 0; i < 4; ++i) acc[i] = vzero8();
  acc[0] = wmma_bf16(aA, bA0, acc[0]);
  acc[1] = wmma_bf16(aA, bA1, acc[1]);
  acc[2] = wmma_bf16(aA, bA2, acc[2]);
  acc[3] = wmma_bf16(aA, bA3, acc[3]);
  acc[0] = wmma_bf16(aB, bB0, acc[0]);
  acc[1] = wmma_bf16(aB, bB1, acc[1]);
  acc[2] = wmma_bf16(aB, bB2, acc[2]);
  acc[3] = wmma_bf16(aB, bB3, acc[3]);
#pragma unroll
  for (int i = 0; i < 4; ++i) {
    const int col = n0 + i * 16 + ml;
    if (col >= SPAD) continue;
#pragma unroll
    for (int r = 0; r < 8; ++r) {
      const int row = mt * 16 + hf * 8 + r;
      scores[((size_t)bh * SPAD + row) * KPAD + col] = acc[i][r] * 0.125f;
    }
  }
}

// In-place softmax: read f32 row, write bf16 probs over the same row bytes
// (stride PSTR), zero-padding cols [197,224) and full pad rows.
__global__ __launch_bounds__(256) void attn_softmax(float* __restrict__ scores) {
  const int bh = blockIdx.x, r = blockIdx.y, tid = threadIdx.x;
  float* row = scores + ((size_t)bh * SPAD + r) * KPAD;
  __bf16* pb = (__bf16*)scores + ((size_t)bh * SPAD + r) * PSTR;
  if (r >= SEQL) {
    if (tid < KPAD) pb[tid] = f2bf(0.0f);
    return;
  }
  __shared__ float red[256];
  float x = (tid < SEQL) ? row[tid] : -3.4e38f;
  red[tid] = x;
  __syncthreads();
  for (int st = 128; st > 0; st >>= 1) {
    if (tid < st) red[tid] = fmaxf(red[tid], red[tid + st]);
    __syncthreads();
  }
  const float mx = red[0];
  __syncthreads();
  float e = (tid < SEQL) ? __expf(x - mx) : 0.0f;
  red[tid] = e;
  __syncthreads();
  for (int st = 128; st > 0; st >>= 1) {
    if (tid < st) red[tid] += red[tid + st];
    __syncthreads();
  }
  const float inv = 1.0f / red[0];
  __syncthreads();  // all f32 reads complete before bf16 overwrite
  if (tid < KPAD) pb[tid] = (tid < SEQL) ? f2bf(e * inv) : f2bf(0.0f);
}

__global__ void transpose_v(const __bf16* __restrict__ qkv,
                            __bf16* __restrict__ vt) {
  const int idx = blockIdx.x * 256 + threadIdx.x;
  if (idx >= BH * HDIM * KPAD) return;
  const int t = idx % KPAD;
  const int d = (idx / KPAD) % HDIM;
  const int bh = idx / (KPAD * HDIM);
  const int b = bh / NHEADS, h = bh % NHEADS;
  vt[idx] = (t < SEQL)
                ? qkv[(size_t)(b * SEQL + t) * 2304 + 1536 + h * HDIM + d]
                : f2bf(0.0f);
}

__global__ __launch_bounds__(32) void attn_av(const float* __restrict__ scores,
                                              const __bf16* __restrict__ vt,
                                              __bf16* __restrict__ O) {
  const int lane = threadIdx.x & 31;
  const int bh = blockIdx.x, mt = blockIdx.y;
  const int b = bh / NHEADS, h = bh % NHEADS;
  const int hf = lane >> 4, ml = lane & 15;
  const __bf16* probs = (const __bf16*)scores;
  const __bf16* arow = probs + ((size_t)bh * SPAD + mt * 16 + ml) * PSTR + hf * 16;
  const __bf16* wp0 = vt + ((size_t)bh * HDIM + (0 * 16 + ml)) * KPAD + hf * 16;
  const __bf16* wp1 = vt + ((size_t)bh * HDIM + (1 * 16 + ml)) * KPAD + hf * 16;
  const __bf16* wp2 = vt + ((size_t)bh * HDIM + (2 * 16 + ml)) * KPAD + hf * 16;
  const __bf16* wp3 = vt + ((size_t)bh * HDIM + (3 * 16 + ml)) * KPAD + hf * 16;
  v8f acc[4];
#pragma unroll
  for (int i = 0; i < 4; ++i) acc[i] = vzero8();
  // software pipeline over K = 224 (probs zero-padded)
  v16bf a = load_a(arow);
  v16bf b0 = load_b16(wp0), b1 = load_b16(wp1);
  v16bf b2 = load_b16(wp2), b3 = load_b16(wp3);
#pragma unroll
  for (int ks = 0; ks < 6; ++ks) {
    arow += 32; wp0 += 32; wp1 += 32; wp2 += 32; wp3 += 32;
    v16bf na = load_a(arow);
    v16bf nb0 = load_b16(wp0), nb1 = load_b16(wp1);
    v16bf nb2 = load_b16(wp2), nb3 = load_b16(wp3);
    acc[0] = wmma_bf16(a, b0, acc[0]);
    acc[1] = wmma_bf16(a, b1, acc[1]);
    acc[2] = wmma_bf16(a, b2, acc[2]);
    acc[3] = wmma_bf16(a, b3, acc[3]);
    a = na; b0 = nb0; b1 = nb1; b2 = nb2; b3 = nb3;
  }
  acc[0] = wmma_bf16(a, b0, acc[0]);
  acc[1] = wmma_bf16(a, b1, acc[1]);
  acc[2] = wmma_bf16(a, b2, acc[2]);
  acc[3] = wmma_bf16(a, b3, acc[3]);
#pragma unroll
  for (int i = 0; i < 4; ++i) {
    const int d = i * 16 + ml;
#pragma unroll
    for (int r = 0; r < 8; ++r) {
      const int s = mt * 16 + hf * 8 + r;
      if (s < SEQL)
        O[(size_t)(b * SEQL + s) * DIMD + h * HDIM + d] = f2bf(acc[i][r]);
    }
  }
}

// ---------------------------------------------------------------------------
// LayerNorm: one block per row; 768 = 3*256 elems; bf16 output.
// ---------------------------------------------------------------------------
__global__ __launch_bounds__(256) void ln_rows(const float* __restrict__ X,
                                               int irow_mult,
                                               const float* __restrict__ g,
                                               const float* __restrict__ bta,
                                               __bf16* __restrict__ out) {
  const int row = blockIdx.x, tid = threadIdx.x;
  const float* xr = X + (size_t)row * irow_mult * DIMD;
  float v0 = xr[tid], v1 = xr[tid + 256], v2 = xr[tid + 512];
  __shared__ float red[256];
  red[tid] = v0 + v1 + v2;
  __syncthreads();
  for (int st = 128; st > 0; st >>= 1) {
    if (tid < st) red[tid] += red[tid + st];
    __syncthreads();
  }
  const float mean = red[0] * (1.0f / DIMD);
  __syncthreads();
  float d0 = v0 - mean, d1 = v1 - mean, d2 = v2 - mean;
  red[tid] = d0 * d0 + d1 * d1 + d2 * d2;
  __syncthreads();
  for (int st = 128; st > 0; st >>= 1) {
    if (tid < st) red[tid] += red[tid + st];
    __syncthreads();
  }
  const float rs = rsqrtf(red[0] * (1.0f / DIMD) + 1e-6f);
  __bf16* orow = out + (size_t)row * DIMD;
  orow[tid]       = f2bf(d0 * rs * g[tid] + bta[tid]);
  orow[tid + 256] = f2bf(d1 * rs * g[tid + 256] + bta[tid + 256]);
  orow[tid + 512] = f2bf(d2 * rs * g[tid + 512] + bta[tid + 512]);
}

__global__ void im2col_k(const float* __restrict__ img, __bf16* __restrict__ Ap) {
  const int idx = blockIdx.x * 256 + threadIdx.x;
  if (idx >= PROWS * DIMD) return;
  const int k = idx % DIMD, row = idx / DIMD;
  const int c = k >> 8, p = (k >> 4) & 15, q = k & 15;
  const int b = row / 196, pp = row % 196;
  const int gh = pp / 14, gw = pp % 14;
  const float v = img[(((size_t)b * 3 + c) * 224 + gh * 16 + p) * 224 + gw * 16 + q];
  Ap[idx] = f2bf(v);
}

__global__ void cls_pos_k(const float* __restrict__ cls_t,
                          const float* __restrict__ pos, float* __restrict__ X) {
  const int idx = blockIdx.x * 256 + threadIdx.x;
  if (idx >= NB * DIMD) return;
  const int b = idx / DIMD, c = idx % DIMD;
  X[(size_t)b * SEQL * DIMD + c] = cls_t[c] + pos[c];
}

// f32 -> bf16 weight pre-conversion, 4 elems/thread (all sizes % 4 == 0).
__global__ void cvt_w4(const float* __restrict__ src, __bf16* __restrict__ dst,
                       int n4) {
  const int i = blockIdx.x * 256 + threadIdx.x;
  if (i >= n4) return;
  v4f f = ((const v4f*)src)[i];
  v4bf r;
#pragma unroll
  for (int j = 0; j < 4; ++j) r[j] = f2bf(f[j]);
  ((v4bf*)dst)[i] = r;
}

// ---------------------------------------------------------------------------
template <int EP>
static void launch_gemm(const __bf16* A, int lda, const __bf16* W, int ldw,
                        const float* bias, float* Cf, __bf16* Cb, int ldc,
                        int Mtiles, int Ngroups, int Nreal, int Ksteps,
                        const float* pos, hipStream_t stream) {
  const int waves = Mtiles * Ngroups;
  const int blocks = (waves + 7) / 8;
  gemm_aw<EP><<<blocks, 256, 0, stream>>>(A, lda, W, ldw, bias, Cf, Cb, ldc,
                                          Mtiles, Ngroups, Nreal, Ksteps, pos);
}

static void launch_cvt(const float* src, __bf16* dst, size_t n,
                       hipStream_t stream) {
  int n4 = (int)(n / 4);
  cvt_w4<<<(n4 + 255) / 256, 256, 0, stream>>>(src, dst, n4);
}

extern "C" void kernel_launch(void* const* d_in, const int* in_sizes, int n_in,
                              void* d_out, int out_size, void* d_ws,
                              size_t ws_size, hipStream_t stream) {
  (void)in_sizes; (void)n_in; (void)out_size; (void)ws_size;
  const float* img    = (const float*)d_in[0];
  const float* conv_w = (const float*)d_in[1];
  const float* conv_b = (const float*)d_in[2];
  const float* cls_t  = (const float*)d_in[3];
  const float* pos    = (const float*)d_in[4];
  const float* ln1_g  = (const float*)d_in[5];
  const float* ln1_b  = (const float*)d_in[6];
  const float* qkv_w  = (const float*)d_in[7];
  const float* qkv_b  = (const float*)d_in[8];
  const float* out_w  = (const float*)d_in[9];
  const float* out_b  = (const float*)d_in[10];
  const float* ln2_g  = (const float*)d_in[11];
  const float* ln2_b  = (const float*)d_in[12];
  const float* w1     = (const float*)d_in[13];
  const float* b1     = (const float*)d_in[14];
  const float* w2     = (const float*)d_in[15];
  const float* b2     = (const float*)d_in[16];
  const float* lnf_g  = (const float*)d_in[17];
  const float* lnf_b  = (const float*)d_in[18];
  const float* fc_w   = (const float*)d_in[19];
  const float* fc_b   = (const float*)d_in[20];

  char* wsp = (char*)d_ws;
  size_t off = 0;
  auto alloc = [&](size_t bytes) -> void* {
    void* p = wsp + off;
    off = (off + bytes + 255) & ~(size_t)255;
    return p;
  };
  float*  X   = (float*)alloc((size_t)NROWS * DIMD * 4);       // residual stream
  __bf16* Hb  = (__bf16*)alloc((size_t)NROWS * DIMD * 2);      // LN output
  __bf16* G   = (__bf16*)alloc((size_t)NROWS * MLPD * 2);      // GELU out / im2col
  __bf16* QKV = (__bf16*)alloc((size_t)NROWS * 2304 * 2);
  __bf16* O   = (__bf16*)alloc((size_t)NROWS * DIMD * 2);      // attn output
  float*  S   = (float*)alloc((size_t)BH * SPAD * KPAD * 4);   // scores/probs
  __bf16* Vt  = (__bf16*)alloc((size_t)BH * HDIM * KPAD * 2);  // V transposed
  __bf16* Ac  = (__bf16*)alloc((size_t)NB * DIMD * 2);         // lnf(CLS rows)
  // bf16 weight staging (per-layer reuse)
  __bf16* WBconv = (__bf16*)alloc((size_t)DIMD * DIMD * 2);
  __bf16* WBfc   = (__bf16*)alloc((size_t)1000 * DIMD * 2);
  __bf16* WBqkv  = (__bf16*)alloc((size_t)2304 * DIMD * 2);
  __bf16* WBout  = (__bf16*)alloc((size_t)DIMD * DIMD * 2);
  __bf16* WBw1   = (__bf16*)alloc((size_t)MLPD * DIMD * 2);
  __bf16* WBw2   = (__bf16*)alloc((size_t)DIMD * MLPD * 2);

  // Patch embedding: im2col (C*P*P == 768 == K) then a 6272x768x768 GEMM.
  {
    int total = PROWS * DIMD;
    im2col_k<<<(total + 255) / 256, 256, 0, stream>>>(img, G);
  }
  {
    int total = NB * DIMD;
    cls_pos_k<<<(total + 255) / 256, 256, 0, stream>>>(cls_t, pos, X);
  }
  launch_cvt(conv_w, WBconv, (size_t)DIMD * DIMD, stream);
  launch_gemm<4>(G, DIMD, WBconv, DIMD, conv_b, X, nullptr, DIMD, PROWS / 32,
                 DIMD / 64, DIMD, DIMD / 32, pos, stream);

  for (int L = 0; L < 12; ++L) {
    launch_cvt(qkv_w + (size_t)L * 2304 * DIMD, WBqkv, (size_t)2304 * DIMD, stream);
    launch_cvt(out_w + (size_t)L * DIMD * DIMD, WBout, (size_t)DIMD * DIMD, stream);
    launch_cvt(w1 + (size_t)L * MLPD * DIMD, WBw1, (size_t)MLPD * DIMD, stream);
    launch_cvt(w2 + (size_t)L * DIMD * MLPD, WBw2, (size_t)DIMD * MLPD, stream);

    ln_rows<<<NROWS, 256, 0, stream>>>(X, 1, ln1_g + L * DIMD, ln1_b + L * DIMD, Hb);
    launch_gemm<1>(Hb, DIMD, WBqkv, DIMD, qkv_b + L * 2304, nullptr, QKV, 2304,
                   NROWS / 32, 2304 / 64, 2304, DIMD / 32, nullptr, stream);
    {
      int total = BH * HDIM * KPAD;
      transpose_v<<<(total + 255) / 256, 256, 0, stream>>>(QKV, Vt);
    }
    attn_qk<<<dim3(BH, SPAD / 16), 128, 0, stream>>>(QKV, S);
    attn_softmax<<<dim3(BH, SPAD), 256, 0, stream>>>(S);
    attn_av<<<dim3(BH, SPAD / 16), 32, 0, stream>>>(S, Vt, O);
    launch_gemm<3>(O, DIMD, WBout, DIMD, out_b + L * DIMD, X, nullptr, DIMD,
                   NROWS / 32, DIMD / 64, DIMD, DIMD / 32, nullptr, stream);
    ln_rows<<<NROWS, 256, 0, stream>>>(X, 1, ln2_g + L * DIMD, ln2_b + L * DIMD, Hb);
    launch_gemm<2>(Hb, DIMD, WBw1, DIMD, b1 + L * MLPD, nullptr, G, MLPD,
                   NROWS / 32, MLPD / 64, MLPD, DIMD / 32, nullptr, stream);
    launch_gemm<3>(G, MLPD, WBw2, MLPD, b2 + L * DIMD, X, nullptr, DIMD,
                   NROWS / 32, DIMD / 64, DIMD, MLPD / 32, nullptr, stream);
  }

  // Final LN on CLS rows, then head GEMM (N=1000 padded to 1024, clamped).
  ln_rows<<<NB, 256, 0, stream>>>(X, SEQL, lnf_g, lnf_b, Ac);
  launch_cvt(fc_w, WBfc, (size_t)1000 * DIMD, stream);
  launch_gemm<5>(Ac, DIMD, WBfc, DIMD, fc_b, (float*)d_out, nullptr, 1000, 1,
                 16, 1000, DIMD / 32, nullptr, stream);
}
